// AttnGlobal_12807592477384
// MI455X (gfx1250) — matrix operand order repaired and checked by reading, not compiled, not run on hardware
//
#include <hip/hip_runtime.h>

typedef __attribute__((ext_vector_type(16))) _Float16 v16h;
typedef __attribute__((ext_vector_type(8)))  float    v8f;

#define BATCH 8
#define SEQ   4096
#define DIM   128
#define ROWS  (BATCH*SEQ)   // 32768
#define NCHUNK (SEQ/64)     // 64

union FragA { v16h h; unsigned u[8]; };
union FragC { v8f  f; float    x[8]; };

__device__ inline v8f wmma_f16(const FragA& a, const FragA& b, v8f c) {
  return __builtin_amdgcn_wmma_f32_16x16x32_f16(
      false, a.h, false, b.h, (short)0, c, false, false);
}

// ---------------------------------------------------------------------------
// CDNA5 async global->LDS copy (GLOBAL_LOAD_ASYNC_TO_LDS_B128, ASYNCcnt),
// with graceful fallback if the builtins are absent on this toolchain.
// ---------------------------------------------------------------------------
#if defined(__AMDGCN__) && __has_builtin(__builtin_amdgcn_global_load_async_to_lds_b128) && __has_builtin(__builtin_amdgcn_s_wait_asynccnt)
#define HAVE_ASYNC_LDS 1
#else
#define HAVE_ASYNC_LDS 0
#endif

#if HAVE_ASYNC_LDS
typedef int b128v __attribute__((vector_size(16)));
typedef __attribute__((address_space(1))) b128v gas_b128;
typedef __attribute__((address_space(3))) b128v las_b128;
__device__ inline void async_copy_b128(const void* g, void* l) {
  __builtin_amdgcn_global_load_async_to_lds_b128(
      (gas_b128*)g, (las_b128*)l, 0, 0);
}
__device__ inline void async_wait() { __builtin_amdgcn_s_wait_asynccnt(0); }
#else
__device__ inline void async_copy_b128(const void* g, void* l) {
  *(uint4*)l = *(const uint4*)g;
}
__device__ inline void async_wait() {}
#endif

// ---------------------------------------------------------------------------
// Elementwise f32 -> f16 (with scale) conversion
// ---------------------------------------------------------------------------
__global__ void cvt_f32_to_f16(const float* __restrict__ src,
                               _Float16* __restrict__ dst, float scale, int n) {
  int i = blockIdx.x * blockDim.x + threadIdx.x;
  if (i < n) dst[i] = (_Float16)(src[i] * scale);
}

// ---------------------------------------------------------------------------
// KV projection: [32768 x 128](f32 x) @ [128 x 256](f16 Wkv) + bkv
// -> k16, v16 each [32768 x 128] f16.  Tile 64(M) x 64(N), 4 waves/WG.
// ---------------------------------------------------------------------------
__global__ __launch_bounds__(128)
void kv_proj_wmma(const float* __restrict__ X, const _Float16* __restrict__ W,
                  const float* __restrict__ bias,
                  _Float16* __restrict__ K, _Float16* __restrict__ V) {
  __shared__ _Float16 ldsA[64 * 136];
  __shared__ _Float16 ldsB[128 * 72];
  const int m0 = blockIdx.x * 64;
  const int n0 = blockIdx.y * 64;
  const int tid = threadIdx.x, lane = tid & 31, wave = tid >> 5;

  // Stage B tile [128 x 64] from W [128 x 256] via async LDS copy (stride 72)
  {
    const uint4* gw = (const uint4*)W;
    for (int i = tid; i < 1024; i += 128) {
      int k = i >> 3, c = i & 7;
      async_copy_b128(&gw[(k * 256 + n0) / 8 + c], &ldsB[k * 72 + c * 8]);
    }
    // Stage A tile [64 x 128] f32 -> f16 in LDS (stride 136), manual convert
    const float4* g = (const float4*)(X + (size_t)m0 * DIM);
    for (int i = tid; i < 2048; i += 128) {
      int r = i >> 5, c = (i & 31) * 4;
      float4 d = g[i];
      union { _Float16 h[4]; uint2 u; } t;
      t.h[0] = (_Float16)d.x; t.h[1] = (_Float16)d.y;
      t.h[2] = (_Float16)d.z; t.h[3] = (_Float16)d.w;
      *(uint2*)(&ldsA[r * 136 + c]) = t.u;
    }
  }
  async_wait();
  __syncthreads();

  FragC acc[4];
  for (int t = 0; t < 4; ++t)
    for (int r = 0; r < 8; ++r) acc[t].x[r] = 0.f;

  const int rbase = wave * 16;
  for (int kc = 0; kc < 4; ++kc) {
    FragA a;
    {
      int row = rbase + (lane & 15);
      int colb = kc * 32 + ((lane >> 4) & 1) * 8;
      const unsigned* p32 = (const unsigned*)ldsA;
      for (int p = 0; p < 8; ++p) {
        int col = colb + ((p & 4) << 2) + ((p & 3) << 1);
        a.u[p] = p32[(row * 136 + col) >> 1];
      }
    }
    for (int t = 0; t < 4; ++t) {
      FragA b;
      const unsigned* p32 = (const unsigned*)ldsB;
      int k = kc * 32 + ((lane >> 4) & 1) * 8; (void)p32;
      for (int s = 0; s < 16; ++s) b.h[s] = ldsB[(k + ((s & 8) << 1) + (s & 7)) * 72 + t * 16 + (lane & 15)];
      acc[t].f = wmma_f16(a, b, acc[t].f);
    }
  }

  for (int t = 0; t < 4; ++t) {
    int col = n0 + t * 16 + (lane & 15);
    float bv = bias[col];
    _Float16* dst = (col < DIM) ? K : V;
    int cc = (col < DIM) ? col : col - DIM;
    for (int r = 0; r < 8; ++r) {
      int row = m0 + rbase + r + ((lane >> 4) & 1) * 8;
      dst[(size_t)row * DIM + cc] = (_Float16)(acc[t].x[r] + bv);
    }
  }
}

// ---------------------------------------------------------------------------
// Flash attention with double-buffered async staging.
// out[n,:] = softmax_m(k[n]·q[m]) @ v[m] per batch; K16 = attention queries.
// ---------------------------------------------------------------------------
__device__ inline void stage_chunk(const _Float16* __restrict__ Q16,
                                   const _Float16* __restrict__ V16,
                                   size_t mrow0, _Float16* qt, _Float16* vb,
                                   int tid) {
  // V chunk [64 m x 128 d] natural layout, async b128 (stride 136)
  const uint4* gv = (const uint4*)(V16 + mrow0 * DIM);
  for (int i = tid; i < 1024; i += 128) {
    int r = i >> 4, c = i & 15;
    async_copy_b128(&gv[i], &vb[r * 136 + c * 8]);
  }
  // Q chunk transposed [d][m] (stride 68), manual scatter
  const uint4* gq = (const uint4*)(Q16 + mrow0 * DIM);
  for (int i = tid; i < 1024; i += 128) {
    int m = i >> 4, c = (i & 15) * 8;
    uint4 d = gq[i];
    const _Float16* h = (const _Float16*)&d;
    for (int e = 0; e < 8; ++e) qt[(c + e) * 68 + m] = h[e];
  }
}

__global__ __launch_bounds__(128)
void flash_attn_wmma(const _Float16* __restrict__ K16,
                     const _Float16* __restrict__ Q16,
                     const _Float16* __restrict__ V16,
                     _Float16* __restrict__ O16) {
  __shared__ _Float16 ldsQt[2][128 * 68];   // Q chunk transposed [d][m]
  __shared__ _Float16 ldsVb[2][64 * 136];   // V chunk [m][d] (buf1 also K stage)
  __shared__ _Float16 ldsP[4 * 16 * 68];    // per-wave P re-layout buffer

  const int tile = blockIdx.x;              // 0..511 (64 rows each)
  const int tid = threadIdx.x, lane = tid & 31, wave = tid >> 5;
  const size_t rowg0 = (size_t)tile * 64;
  const size_t batch0 = (size_t)((tile * 64) / SEQ) * SEQ;
  const int half = (lane >> 4) & 1, cl = lane & 15;

  // Stage this WG's K rows (async) into buffer 1's V region
  {
    const uint4* gk = (const uint4*)(K16 + rowg0 * DIM);
    for (int i = tid; i < 1024; i += 128) {
      int r = i >> 4, c = i & 15;
      async_copy_b128(&gk[i], &ldsVb[1][r * 136 + c * 8]);
    }
  }
  async_wait();
  __syncthreads();

  // Persistent A fragments of K rows (16x32 x 4 d-chunks per wave)
  FragA aK[4];
  {
    int row = wave * 16 + cl;
    int colb0 = half * 8;
    const unsigned* p32 = (const unsigned*)&ldsVb[1][0];
    for (int kc = 0; kc < 4; ++kc)
      for (int p = 0; p < 8; ++p) {
        int col = kc * 32 + colb0 + ((p & 4) << 2) + ((p & 3) << 1);
        aK[kc].u[p] = p32[(row * 136 + col) >> 1];
      }
  }
  __syncthreads();

  // Prologue: stage chunk 0 into buffer 0
  stage_chunk(Q16, V16, batch0, &ldsQt[0][0], &ldsVb[0][0], tid);

  FragC o[8];
  for (int t = 0; t < 8; ++t)
    for (int r = 0; r < 8; ++r) o[t].x[r] = 0.f;
  float rmax[8], rsum[8];
  for (int r = 0; r < 8; ++r) { rmax[r] = -1e30f; rsum[r] = 0.f; }

  for (int j = 0; j < NCHUNK; ++j) {
    async_wait();
    __syncthreads();
    const int cur = j & 1;
    const _Float16* qt = &ldsQt[cur][0];
    const _Float16* vb = &ldsVb[cur][0];

    // Kick off next chunk's staging into the other buffer while we compute
    if (j + 1 < NCHUNK) {
      stage_chunk(Q16, V16, batch0 + (size_t)(j + 1) * 64,
                  &ldsQt[1 - cur][0], &ldsVb[1 - cur][0], tid);
      if (j + 2 < NCHUNK) {
        const char* pq =
            (const char*)(Q16 + (batch0 + (size_t)(j + 2) * 64) * DIM) + tid * 128;
        const char* pv =
            (const char*)(V16 + (batch0 + (size_t)(j + 2) * 64) * DIM) + tid * 128;
        __builtin_prefetch(pq, 0, 1);
        __builtin_prefetch(pv, 0, 1);
      }
    }

    // Scores S[16 x 64] = K_rows @ Q_chunk^T  (16 WMMAs/wave)
    FragC s[4];
    for (int t = 0; t < 4; ++t)
      for (int r = 0; r < 8; ++r) s[t].x[r] = 0.f;
    for (int kc = 0; kc < 4; ++kc) {
      for (int t = 0; t < 4; ++t) {
        FragA bq;
        const unsigned* p32 = (const unsigned*)qt;
        int d = kc * 32 + half * 8; (void)p32;
        for (int s = 0; s < 16; ++s)
          bq.h[s] = qt[(d + ((s & 8) << 1) + (s & 7)) * 68 + t * 16 + cl];
        s[t].f = wmma_f16(aK[kc], bq, s[t].f);
      }
    }

    // Online softmax (stats per (reg, lane-half); reduce over 16 lanes)
    for (int r = 0; r < 8; ++r) {
      float mx = fmaxf(fmaxf(s[0].x[r], s[1].x[r]), fmaxf(s[2].x[r], s[3].x[r]));
      for (int sh = 8; sh >= 1; sh >>= 1) mx = fmaxf(mx, __shfl_xor(mx, sh, 32));
      float nm = fmaxf(rmax[r], mx);
      float alpha = __expf(rmax[r] - nm);
      float rs = 0.f;
      for (int t = 0; t < 4; ++t) {
        float p = __expf(s[t].x[r] - nm);
        s[t].x[r] = p;
        rs += p;
      }
      for (int sh = 8; sh >= 1; sh >>= 1) rs += __shfl_xor(rs, sh, 32);
      rmax[r] = nm;
      rsum[r] = rsum[r] * alpha + rs;
      for (int nt = 0; nt < 8; ++nt) o[nt].x[r] *= alpha;
    }

    // Re-layout P (C-frag -> A-frag) through per-wave LDS
    {
      _Float16* pw = &ldsP[wave * 16 * 68];
      for (int t = 0; t < 4; ++t)
        for (int r = 0; r < 8; ++r)
          pw[(r + half * 8) * 68 + t * 16 + cl] = (_Float16)s[t].x[r];
    }
    FragA aP[2];
    {
      const unsigned* p32 = (const unsigned*)&ldsP[wave * 16 * 68];
      int row = cl, colb0 = half * 8;
      for (int kc = 0; kc < 2; ++kc)
        for (int p = 0; p < 8; ++p) {
          int col = kc * 32 + colb0 + ((p & 4) << 2) + ((p & 3) << 1);
          aP[kc].u[p] = p32[(row * 68 + col) >> 1];
        }
    }

    // O += P @ V_chunk  (16 WMMAs/wave)
    for (int kc = 0; kc < 2; ++kc) {
      for (int nt = 0; nt < 8; ++nt) {
        FragA bv;
        const unsigned* p32 = (const unsigned*)vb;
        int m = kc * 32 + half * 8; (void)p32;
        for (int s = 0; s < 16; ++s)
          bv.h[s] = vb[(m + ((s & 8) << 1) + (s & 7)) * 136 + nt * 16 + cl];
        o[nt].f = wmma_f16(aP[kc], bv, o[nt].f);
      }
    }
    __syncthreads();
  }

  // Normalize and store f16 attention output
  for (int r = 0; r < 8; ++r) {
    float inv = 1.f / rsum[r];
    size_t row = rowg0 + wave * 16 + r + half * 8;
    for (int nt = 0; nt < 8; ++nt)
      O16[row * DIM + nt * 16 + cl] = (_Float16)(o[nt].x[r] * inv);
  }
}

// ---------------------------------------------------------------------------
// Output projection: [32768 x 128](f16) @ [128 x 128](f16 Wp) + bp -> f32 out
// ---------------------------------------------------------------------------
__global__ __launch_bounds__(128)
void out_proj_wmma(const _Float16* __restrict__ A16,
                   const _Float16* __restrict__ Wp16,
                   const float* __restrict__ bp, float* __restrict__ out) {
  __shared__ _Float16 ldsA[64 * 136];
  __shared__ _Float16 ldsB[128 * 72];
  const int m0 = blockIdx.x * 64;
  const int n0 = blockIdx.y * 64;
  const int tid = threadIdx.x, lane = tid & 31, wave = tid >> 5;

  {
    const uint4* g = (const uint4*)(A16 + (size_t)m0 * DIM);
    for (int i = tid; i < 1024; i += 128) {
      int r = i >> 4, c = i & 15;
      async_copy_b128(&g[i], &ldsA[r * 136 + c * 8]);
    }
    const uint4* gw = (const uint4*)Wp16;
    for (int i = tid; i < 1024; i += 128) {
      int k = i >> 3, c = i & 7;
      async_copy_b128(&gw[(k * 128 + n0) / 8 + c], &ldsB[k * 72 + c * 8]);
    }
  }
  async_wait();
  __syncthreads();

  FragC acc[4];
  for (int t = 0; t < 4; ++t)
    for (int r = 0; r < 8; ++r) acc[t].x[r] = 0.f;

  const int rbase = wave * 16;
  for (int kc = 0; kc < 4; ++kc) {
    FragA a;
    {
      int row = rbase + (lane & 15);
      int colb = kc * 32 + ((lane >> 4) & 1) * 8;
      const unsigned* p32 = (const unsigned*)ldsA;
      for (int p = 0; p < 8; ++p) {
        int col = colb + ((p & 4) << 2) + ((p & 3) << 1);
        a.u[p] = p32[(row * 136 + col) >> 1];
      }
    }
    for (int t = 0; t < 4; ++t) {
      FragA b;
      const unsigned* p32 = (const unsigned*)ldsB;
      int k = kc * 32 + ((lane >> 4) & 1) * 8; (void)p32;
      for (int s = 0; s < 16; ++s) b.h[s] = ldsB[(k + ((s & 8) << 1) + (s & 7)) * 72 + t * 16 + (lane & 15)];
      acc[t].f = wmma_f16(a, b, acc[t].f);
    }
  }

  for (int t = 0; t < 4; ++t) {
    int col = n0 + t * 16 + (lane & 15);
    float bv = bp[col];
    for (int r = 0; r < 8; ++r) {
      int row = m0 + rbase + r + ((lane >> 4) & 1) * 8;
      out[(size_t)row * DIM + col] = acc[t].x[r] + bv;
    }
  }
}

// ---------------------------------------------------------------------------
extern "C" void kernel_launch(void* const* d_in, const int* in_sizes, int n_in,
                              void* d_out, int out_size, void* d_ws, size_t ws_size,
                              hipStream_t stream) {
  (void)in_sizes; (void)n_in; (void)out_size; (void)ws_size;
  const float* x   = (const float*)d_in[0];
  const float* qg  = (const float*)d_in[1];
  const float* Wkv = (const float*)d_in[2];
  const float* bkv = (const float*)d_in[3];
  const float* Wp  = (const float*)d_in[4];
  const float* bp  = (const float*)d_in[5];
  float* out = (float*)d_out;

  char* ws = (char*)d_ws;
  size_t off = 0;
  auto alloc = [&](size_t bytes) -> void* {
    void* p = ws + off;
    off += (bytes + 255) & ~(size_t)255;
    return p;
  };
  _Float16* q16   = (_Float16*)alloc((size_t)ROWS * DIM * 2);
  _Float16* k16   = (_Float16*)alloc((size_t)ROWS * DIM * 2);
  _Float16* v16   = (_Float16*)alloc((size_t)ROWS * DIM * 2);
  _Float16* p16   = (_Float16*)alloc((size_t)ROWS * DIM * 2);
  _Float16* wkv16 = (_Float16*)alloc((size_t)DIM * 2 * DIM * 2);
  _Float16* wp16  = (_Float16*)alloc((size_t)DIM * DIM * 2);

  const int nElem = ROWS * DIM;
  const float qscale = 0.08838834764831845f;  // 1/sqrt(128)
  cvt_f32_to_f16<<<(nElem + 255) / 256, 256, 0, stream>>>(qg, q16, qscale, nElem);
  cvt_f32_to_f16<<<(DIM * 2 * DIM + 255) / 256, 256, 0, stream>>>(Wkv, wkv16, 1.f, DIM * 2 * DIM);
  cvt_f32_to_f16<<<(DIM * DIM + 255) / 256, 256, 0, stream>>>(Wp, wp16, 1.f, DIM * DIM);

  dim3 g1(ROWS / 64, 4);
  kv_proj_wmma<<<g1, 128, 0, stream>>>(x, wkv16, bkv, k16, v16);

  flash_attn_wmma<<<ROWS / 64, 128, 0, stream>>>(k16, q16, v16, p16);

  dim3 g3(ROWS / 64, 2);
  out_proj_wmma<<<g3, 128, 0, stream>>>(p16, wp16, bp, out);
}
